// GRU_with_cells_category_edited_86268713108352
// MI455X (gfx1250) — matrix operand order repaired and checked by reading, not compiled, not run on hardware
//
#include <hip/hip_runtime.h>

// ---------------------------------------------------------------------------
// CDNA5 (gfx1250) GRU: bf16 WMMA GEMMs (async-to-LDS double-buffered) +
// persistent grid-synced recurrence.  B=16, T=128, H=512, V=32000.
// ---------------------------------------------------------------------------

typedef __attribute__((ext_vector_type(16))) __bf16 v16bf;
typedef __attribute__((ext_vector_type(8)))  float  v8f;
typedef __attribute__((ext_vector_type(4)))  int    v4i;

#define HB 512
#define H3 1536
#define BT 2048          // B*T rows
#define NBLK_REC 24      // persistent blocks in recurrent kernel

// ---- async global->LDS copy (16 bytes per lane), ASYNCcnt tracked ----------
__device__ __forceinline__ void async_copy_b128(const __bf16* gptr, __bf16* lptr) {
#if __has_builtin(__builtin_amdgcn_global_load_async_to_lds_b128)
  __attribute__((address_space(3))) __bf16* l3 =
      (__attribute__((address_space(3))) __bf16*)lptr;
  __builtin_amdgcn_global_load_async_to_lds_b128(
      (__attribute__((address_space(1))) v4i*)(unsigned long long)gptr,
      (__attribute__((address_space(3))) v4i*)l3, 0, 0);
#else
  unsigned loff = (unsigned)(unsigned long long)
      (__attribute__((address_space(3))) __bf16*)lptr;
  unsigned long long ga = (unsigned long long)gptr;
  asm volatile("global_load_async_to_lds_b128 %0, %1, off"
               :: "v"(loff), "v"(ga) : "memory");
#endif
}

__device__ __forceinline__ void wait_async0() {
#if __has_builtin(__builtin_amdgcn_s_wait_asynccnt)
  __builtin_amdgcn_s_wait_asynccnt(0);
#else
  asm volatile("s_wait_asynccnt 0x0" ::: "memory");
#endif
}

// ---- WMMA fragment loaders -------------------------------------------------
// A (16x32 bf16): lane L(0-15) holds M=L, K = {0..7, 16..23}; lanes 16-31 hold
// M=L-16, K = {8..15, 24..31}.  Two contiguous 16B loads per lane.
__device__ __forceinline__ v16bf load_fragA(const __bf16* row, int k0, int half) {
  union { v16bf v; uint4 q[2]; } u;
  u.q[0] = *reinterpret_cast<const uint4*>(row + k0 + (half << 3));
  u.q[1] = *reinterpret_cast<const uint4*>(row + k0 + 16 + (half << 3));
  return u.v;
}
// B (32x16 bf16, column-major source W[n, k]): lane L(0-15) holds N=L, K = {0..7, 16..23};
// lanes 16-31 hold K = {8..15, 24..31}.  Two contiguous 16B loads per lane.
__device__ __forceinline__ v16bf load_fragB(const __bf16* row, int k0, int half) {
  union { v16bf v; uint4 q[2]; } u;
  const __bf16* p = row + k0 + (half << 3);
  u.q[0] = *reinterpret_cast<const uint4*>(p);
  u.q[1] = *reinterpret_cast<const uint4*>(p + 16);
  return u.v;
}

__device__ __forceinline__ float sigm(float x) { return 1.0f / (1.0f + __expf(-x)); }

// ---- device-wide split barrier (all NBLK_REC blocks resident) --------------
__device__ __forceinline__ void gsync(unsigned* sync, unsigned nb) {
  __syncthreads();
  if (threadIdx.x == 0) {
    __threadfence();
    unsigned* cnt = sync;
    unsigned* gen = sync + 1;
    unsigned g = __hip_atomic_load(gen, __ATOMIC_RELAXED, __HIP_MEMORY_SCOPE_AGENT);
    unsigned a = __hip_atomic_fetch_add(cnt, 1u, __ATOMIC_ACQ_REL, __HIP_MEMORY_SCOPE_AGENT);
    if (a == nb - 1u) {
      __hip_atomic_store(cnt, 0u, __ATOMIC_RELAXED, __HIP_MEMORY_SCOPE_AGENT);
      __hip_atomic_store(gen, g + 1u, __ATOMIC_RELEASE, __HIP_MEMORY_SCOPE_AGENT);
    } else {
      while (__hip_atomic_load(gen, __ATOMIC_ACQUIRE, __HIP_MEMORY_SCOPE_AGENT) == g) {
        __builtin_amdgcn_s_sleep(1);
      }
    }
    __threadfence();
  }
  __syncthreads();
}

// ---- one 16x16 output tile, K=512, A in LDS (16x512 bf16), W in global -----
// Two independent accumulator chains (even/odd K chunks) to double XDL ILP on
// the latency-critical recurrent path.
__device__ __forceinline__ void mm_tile_512(const __bf16* sA, const __bf16* W, int ldw,
                                            float* outbuf, int n0, int idx, int half) {
  v8f acc0 = {}, acc1 = {};
  const __bf16* wrow = W + (size_t)(n0 + idx) * ldw;
  const __bf16* arow = sA + idx * HB;
#pragma unroll
  for (int k = 0; k < HB; k += 64) {
    v16bf a0 = load_fragA(arow, k, half);
    v16bf b0 = load_fragB(wrow, k, half);
    v16bf a1 = load_fragA(arow, k + 32, half);
    v16bf b1 = load_fragB(wrow, k + 32, half);
    acc0 = __builtin_amdgcn_wmma_f32_16x16x32_bf16(false, a0, false, b0, (short)0, acc0, false, false);
    acc1 = __builtin_amdgcn_wmma_f32_16x16x32_bf16(false, a1, false, b1, (short)0, acc1, false, false);
  }
  v8f acc = acc0 + acc1;
#pragma unroll
  for (int r = 0; r < 8; ++r)
    outbuf[(size_t)((half << 3) + r) * H3 + n0 + idx] = acc[r];
}

// ---------------------------------------------------------------------------
// Generic bf16 WMMA GEMM: out[m, n] = sum_k A[m*lda+k] * W[n*ldw+k] + bias[n]
// Block: 64 M-rows x 64 N-cols (4 waves, each 4 M-tiles of one N-tile).
// A tile staged via async-to-LDS DMA, double-buffered, 1 barrier per K-step.
// permute==1 stores with m=(t*16+b) remapped to row (b*128+t)  [FC output].
// ---------------------------------------------------------------------------
__global__ void __launch_bounds__(128) gemm_bf16_kernel(
    const __bf16* __restrict__ A, int lda,
    const __bf16* __restrict__ W, int ldw,
    const float* __restrict__ bias,
    float* __restrict__ out, int ldo, int K, int permute) {
  __shared__ __bf16 sA[2][64 * 32];
  const int tid  = threadIdx.x;
  const int wave = tid >> 5, lane = tid & 31;
  const int half = lane >> 4, idx = lane & 15;
  const int m0 = blockIdx.x * 64;
  const int n0 = blockIdx.y * 64 + wave * 16;
  v8f acc[4] = {};
  const __bf16* wrow = W + (size_t)(n0 + idx) * ldw;

  // per-thread staging of a 64x32 bf16 tile: 256 x 16B chunks, 2 per thread
  auto stage = [&](int kc, int b) {
#pragma unroll
    for (int u = 0; u < 2; ++u) {
      int chunk = tid + u * 128;
      int r = chunk >> 2;
      int c = (chunk & 3) * 8;
      async_copy_b128(&A[(size_t)(m0 + r) * lda + kc + c], &sA[b][r * 32 + c]);
    }
  };

  const int nk = K >> 5;
  stage(0, 0);
  int buf = 0;
  for (int ki = 0; ki < nk; ++ki) {
    wait_async0();        // this wave's issued copies (tile ki) landed in LDS
    __syncthreads();      // all waves' copies visible; prev reads of buf^1 done
    if (ki + 1 < nk) stage((ki + 1) << 5, buf ^ 1);   // overlap copy w/ compute
    v16bf bf = load_fragB(wrow, ki << 5, half);
    v16bf af0 = load_fragA(&sA[buf][(0 * 16 + idx) * 32], 0, half);
    v16bf af1 = load_fragA(&sA[buf][(1 * 16 + idx) * 32], 0, half);
    v16bf af2 = load_fragA(&sA[buf][(2 * 16 + idx) * 32], 0, half);
    v16bf af3 = load_fragA(&sA[buf][(3 * 16 + idx) * 32], 0, half);
    acc[0] = __builtin_amdgcn_wmma_f32_16x16x32_bf16(false, af0, false, bf, (short)0, acc[0], false, false);
    acc[1] = __builtin_amdgcn_wmma_f32_16x16x32_bf16(false, af1, false, bf, (short)0, acc[1], false, false);
    acc[2] = __builtin_amdgcn_wmma_f32_16x16x32_bf16(false, af2, false, bf, (short)0, acc[2], false, false);
    acc[3] = __builtin_amdgcn_wmma_f32_16x16x32_bf16(false, af3, false, bf, (short)0, acc[3], false, false);
    buf ^= 1;
  }
  const float bv = bias ? bias[n0 + idx] : 0.0f;
#pragma unroll
  for (int mt = 0; mt < 4; ++mt) {
#pragma unroll
    for (int r = 0; r < 8; ++r) {
      int m = m0 + mt * 16 + (half << 3) + r;
      int n = n0 + idx;
      size_t row = permute ? (size_t)((m & 15) * 128 + (m >> 4)) : (size_t)m;
      out[row * (size_t)ldo + n] = acc[mt][r] + bv;
    }
  }
}

// ---------------------------------------------------------------------------
__global__ void cvt_f32_bf16(const float* __restrict__ in, __bf16* __restrict__ out, int n) {
  int i = blockIdx.x * blockDim.x + threadIdx.x;
  if (i < n) out[i] = (__bf16)in[i];
}

// One block per (t,b) row: gathers word+cat embeddings into bf16 buffers.
__global__ void embed_kernel(const int* __restrict__ tok, const int* __restrict__ cat,
                             const float* __restrict__ emb_word, const float* __restrict__ emb_cat,
                             __bf16* __restrict__ comb, __bf16* __restrict__ xcat) {
  int row = blockIdx.x;               // row = t*16 + b
  int t = row >> 4, b = row & 15;
  int tk = tok[b * 128 + t];
  int ct = cat[b * 128 + t];
  const float* we = emb_word + (size_t)tk * HB;
  const float* ce = emb_cat + (size_t)ct * HB;
  for (int j = threadIdx.x; j < HB; j += blockDim.x) {
    __bf16 wv = (__bf16)we[j];
    __bf16 cv = (__bf16)ce[j];
    comb[(size_t)row * 1024 + j]      = wv;
    comb[(size_t)row * 1024 + HB + j] = cv;
    xcat[(size_t)row * 1024 + j]      = cv;   // first half of FC input
  }
}

__global__ void init_state_kernel(const float* h1, const float* h2, const float* h3,
                                  float* h1s, float* h2s, float* h3s, unsigned* sync) {
  int i = blockIdx.x * blockDim.x + threadIdx.x;
  if (i < 8192) { h1s[i] = h1[i]; h2s[i] = h2[i]; h3s[i] = h3[i]; }
  if (i < 2) sync[i] = 0u;
}

// ---------------------------------------------------------------------------
// Persistent recurrent kernel: 24 blocks x 128 threads, 3 grid syncs / step.
// ---------------------------------------------------------------------------
__global__ void __launch_bounds__(128) gru_recurrent_kernel(
    const float* __restrict__ gi1, const float* __restrict__ gi2c, const float* __restrict__ gi3c,
    const __bf16* __restrict__ w_hh1b, const __bf16* __restrict__ w_ih2b,
    const __bf16* __restrict__ w_hh2b, const __bf16* __restrict__ w_ih3b,
    const __bf16* __restrict__ w_hh3b,
    const float* __restrict__ b_hh1, const float* __restrict__ b_hh2, const float* __restrict__ b_hh3,
    float* gh1buf, float* gi2dyn, float* gh2buf, float* gi3dyn, float* gh3buf,
    float* h1s, float* h2s, float* h3s,
    __bf16* xcat, float* out_states, unsigned* sync) {
  __shared__ __bf16 sA0[16 * HB];   // current layer input (bf16 A operand)
  __shared__ __bf16 sA1[16 * HB];   // current hidden state (bf16 A operand)
  const int tid  = threadIdx.x;
  const int wave = tid >> 5, lane = tid & 31;
  const int half = lane >> 4, idx = lane & 15;
  const unsigned nb = gridDim.x;

  for (int t = 0; t < 128; ++t) {
    const int p = t & 1;
    const float* h1r = h1s + p * 8192;  float* h1w = h1s + (p ^ 1) * 8192;
    const float* h2r = h2s + p * 8192;  float* h2w = h2s + (p ^ 1) * 8192;
    const float* h3r = h3s + p * 8192;  float* h3w = h3s + (p ^ 1) * 8192;
    const float* gi1t  = gi1  + (size_t)t * 16 * H3;
    const float* gi2ct = gi2c + (size_t)t * 16 * H3;
    const float* gi3ct = gi3c + (size_t)t * 16 * H3;

    // ---- phase 1: gh1 = h1 @ w_hh1^T ------------------------------------
    for (int i = tid; i < 8192; i += 128) sA0[i] = (__bf16)h1r[i];
    __syncthreads();
    mm_tile_512(sA0, w_hh1b, HB, gh1buf, (blockIdx.x * 4 + wave) * 16, idx, half);
    gsync(sync, nb);

    // ---- phase 2: c1 = gate(gi1, gh1);  gi2dyn = c1@w_ih2h^T; gh2 = h2@w_hh2^T
    for (int i = tid; i < 8192; i += 128) {
      int m = i >> 9, j = i & 511;
      const float* gm = gi1t + m * H3;
      const float* hm = gh1buf + m * H3;
      float r = sigm(gm[j] + hm[j] + b_hh1[j]);
      float z = sigm(gm[HB + j] + hm[HB + j] + b_hh1[HB + j]);
      float n = tanhf(gm[1024 + j] + r * (hm[1024 + j] + b_hh1[1024 + j]));
      float c = (1.0f - z) * n + z * h1r[i];
      sA0[i] = (__bf16)c;
      if (blockIdx.x == 0) h1w[i] = c;
      sA1[i] = (__bf16)h2r[i];
    }
    __syncthreads();
    {
      int tt = (blockIdx.x * 4 + wave) * 2;     // 192 tiles across 96 waves
#pragma unroll
      for (int u = 0; u < 2; ++u) {
        int tile = tt + u;
        if (tile < 96) mm_tile_512(sA0, w_ih2b + HB, 1024, gi2dyn, tile * 16, idx, half);
        else           mm_tile_512(sA1, w_hh2b, HB, gh2buf, (tile - 96) * 16, idx, half);
      }
    }
    gsync(sync, nb);

    // ---- phase 3: c2 = gate(gi2c+gi2dyn, gh2); gi3dyn = c2@w_ih3h^T; gh3 = h3@w_hh3^T
    for (int i = tid; i < 8192; i += 128) {
      int m = i >> 9, j = i & 511;
      const float* gm = gi2ct + m * H3;
      const float* dm = gi2dyn + m * H3;
      const float* hm = gh2buf + m * H3;
      float r = sigm(gm[j] + dm[j] + hm[j] + b_hh2[j]);
      float z = sigm(gm[HB + j] + dm[HB + j] + hm[HB + j] + b_hh2[HB + j]);
      float n = tanhf(gm[1024 + j] + dm[1024 + j] + r * (hm[1024 + j] + b_hh2[1024 + j]));
      float c = (1.0f - z) * n + z * h2r[i];
      sA0[i] = (__bf16)c;
      if (blockIdx.x == 1) h2w[i] = c;
      sA1[i] = (__bf16)h3r[i];
    }
    __syncthreads();
    {
      int tt = (blockIdx.x * 4 + wave) * 2;
#pragma unroll
      for (int u = 0; u < 2; ++u) {
        int tile = tt + u;
        if (tile < 96) mm_tile_512(sA0, w_ih3b + HB, 1024, gi3dyn, tile * 16, idx, half);
        else           mm_tile_512(sA1, w_hh3b, HB, gh3buf, (tile - 96) * 16, idx, half);
      }
    }
    gsync(sync, nb);

    // ---- phase 4 (block 0 only): c3 gate; feed FC input + next h3 --------
    if (blockIdx.x == 0) {
      for (int i = tid; i < 8192; i += 128) {
        int m = i >> 9, j = i & 511;
        const float* gm = gi3ct + m * H3;
        const float* dm = gi3dyn + m * H3;
        const float* hm = gh3buf + m * H3;
        float r = sigm(gm[j] + dm[j] + hm[j] + b_hh3[j]);
        float z = sigm(gm[HB + j] + dm[HB + j] + hm[HB + j] + b_hh3[HB + j]);
        float n = tanhf(gm[1024 + j] + dm[1024 + j] + r * (hm[1024 + j] + b_hh3[1024 + j]));
        float c = (1.0f - z) * n + z * h3r[i];
        h3w[i] = c;
        xcat[((size_t)t * 16 + m) * 1024 + HB + j] = (__bf16)c;   // second half of FC input
      }
    }
    // no sync needed: next consumers are separated by >=2 gsyncs
  }

  // final states live in slot 0 after t=127 (writers copy their own data)
  if (blockIdx.x == 0) {
    for (int i = tid; i < 8192; i += 128) {
      out_states[i]         = h1s[i];
      out_states[16384 + i] = h3s[i];
    }
  } else if (blockIdx.x == 1) {
    for (int i = tid; i < 8192; i += 128) out_states[8192 + i] = h2s[i];
  }
}

// ---------------------------------------------------------------------------
extern "C" void kernel_launch(void* const* d_in, const int* in_sizes, int n_in,
                              void* d_out, int out_size, void* d_ws, size_t ws_size,
                              hipStream_t stream) {
  const int*   tok      = (const int*)d_in[0];
  const float* h1in     = (const float*)d_in[1];
  const float* h2in     = (const float*)d_in[2];
  const float* h3in     = (const float*)d_in[3];
  const int*   cat      = (const int*)d_in[5];
  const float* emb_word = (const float*)d_in[6];
  const float* emb_cat  = (const float*)d_in[7];
  const float* w_ih1 = (const float*)d_in[8];
  const float* w_hh1 = (const float*)d_in[9];
  const float* b_ih1 = (const float*)d_in[10];
  const float* b_hh1 = (const float*)d_in[11];
  const float* w_ih2 = (const float*)d_in[12];
  const float* w_hh2 = (const float*)d_in[13];
  const float* b_ih2 = (const float*)d_in[14];
  const float* b_hh2 = (const float*)d_in[15];
  const float* w_ih3 = (const float*)d_in[16];
  const float* w_hh3 = (const float*)d_in[17];
  const float* b_ih3 = (const float*)d_in[18];
  const float* b_hh3 = (const float*)d_in[19];
  const float* fc_w  = (const float*)d_in[20];
  const float* fc_b  = (const float*)d_in[21];
  float* out = (float*)d_out;
  float* out_states = out + (size_t)16 * 128 * 32000;

  // ---- workspace partition ----
  char* wptr = (char*)d_ws;
  auto alloc = [&](size_t bytes) {
    void* p = (void*)wptr;
    wptr += (bytes + 255) & ~(size_t)255;
    return p;
  };
  __bf16* comb   = (__bf16*)alloc((size_t)BT * 1024 * 2);
  __bf16* xcat   = (__bf16*)alloc((size_t)BT * 1024 * 2);
  float*  gi1    = (float*)alloc((size_t)BT * H3 * 4);
  float*  gi2c   = (float*)alloc((size_t)BT * H3 * 4);
  float*  gi3c   = (float*)alloc((size_t)BT * H3 * 4);
  __bf16* wih1b  = (__bf16*)alloc((size_t)H3 * 1024 * 2);
  __bf16* wih2b  = (__bf16*)alloc((size_t)H3 * 1024 * 2);
  __bf16* wih3b  = (__bf16*)alloc((size_t)H3 * 1024 * 2);
  __bf16* whh1b  = (__bf16*)alloc((size_t)H3 * HB * 2);
  __bf16* whh2b  = (__bf16*)alloc((size_t)H3 * HB * 2);
  __bf16* whh3b  = (__bf16*)alloc((size_t)H3 * HB * 2);
  __bf16* fcwb   = (__bf16*)alloc((size_t)32000 * 1024 * 2);
  float*  gh1buf = (float*)alloc((size_t)16 * H3 * 4);
  float*  gi2dyn = (float*)alloc((size_t)16 * H3 * 4);
  float*  gh2buf = (float*)alloc((size_t)16 * H3 * 4);
  float*  gi3dyn = (float*)alloc((size_t)16 * H3 * 4);
  float*  gh3buf = (float*)alloc((size_t)16 * H3 * 4);
  float*  h1s    = (float*)alloc((size_t)2 * 8192 * 4);
  float*  h2s    = (float*)alloc((size_t)2 * 8192 * 4);
  float*  h3s    = (float*)alloc((size_t)2 * 8192 * 4);
  unsigned* sync = (unsigned*)alloc(256);

  auto cvt = [&](const float* src, __bf16* dst, int n) {
    cvt_f32_bf16<<<(n + 255) / 256, 256, 0, stream>>>(src, dst, n);
  };
  cvt(w_ih1, wih1b, H3 * 1024);
  cvt(w_ih2, wih2b, H3 * 1024);
  cvt(w_ih3, wih3b, H3 * 1024);
  cvt(w_hh1, whh1b, H3 * HB);
  cvt(w_hh2, whh2b, H3 * HB);
  cvt(w_hh3, whh3b, H3 * HB);
  cvt(fc_w,  fcwb,  32000 * 1024);

  embed_kernel<<<BT, 128, 0, stream>>>(tok, cat, emb_word, emb_cat, comb, xcat);
  init_state_kernel<<<64, 128, 0, stream>>>(h1in, h2in, h3in, h1s, h2s, h3s, sync);

  // gi1 = comb @ w_ih1^T + b_ih1           [2048,1024]x[1024,1536]
  gemm_bf16_kernel<<<dim3(32, 24), 128, 0, stream>>>(comb, 1024, wih1b, 1024, b_ih1, gi1, H3, 1024, 0);
  // gi2_cat = ec @ w_ih2[:, :512]^T + b_ih2  (ec = first half of xcat rows)
  gemm_bf16_kernel<<<dim3(32, 24), 128, 0, stream>>>(xcat, 1024, wih2b, 1024, b_ih2, gi2c, H3, 512, 0);
  gemm_bf16_kernel<<<dim3(32, 24), 128, 0, stream>>>(xcat, 1024, wih3b, 1024, b_ih3, gi3c, H3, 512, 0);

  gru_recurrent_kernel<<<NBLK_REC, 128, 0, stream>>>(
      gi1, gi2c, gi3c,
      whh1b, wih2b, whh2b, wih3b, whh3b,
      b_hh1, b_hh2, b_hh3,
      gh1buf, gi2dyn, gh2buf, gi3dyn, gh3buf,
      h1s, h2s, h3s, xcat, out_states, sync);

  // final_output[b,t,:] = concat(ec, c3) @ fc_w^T + fc_b   (permuted store)
  gemm_bf16_kernel<<<dim3(32, 500), 128, 0, stream>>>(xcat, 1024, fcwb, 1024, fc_b, out, 32000, 1024, 1);
}
